// SDFORMERModel_89240830476459
// MI455X (gfx1250) — matrix operand order repaired and checked by reading, not compiled, not run on hardware
//
#include <hip/hip_runtime.h>
#include <cstddef>
#include <cstdint>

// ============================================================================
// SDFormer (Restormer-style U-Net) forward for MI455X / gfx1250, wave32.
//
// Pointwise (1x1) convs = the FLOP-dominant ops -> v_wmma_f32_16x16x32_bf16
// with f32 accumulate. Weights pre-packed per launch into WMMA A-fragment
// layout (bf16, zero-padded); activations zero-padded to K%32==0; NPIX is a
// template constant so B-fragment loads use immediate offsets. Each wave
// accumulates MTILE M-tiles (B-fragment reuse, fewer L2 re-reads), A
// fragments load as one clause, B is double-buffered across K-tiles, and
// global_prefetch covers the tile after next. Border taps in the 3x3 convs
// use clamp+select (no EXEC divergence). Working set fits in the 192MB L2.
// ============================================================================

typedef __attribute__((ext_vector_type(16))) __bf16 v16bf;
typedef __attribute__((ext_vector_type(8)))  float  v8f;

// ---------------------------------------------------------------------------
// conv3x3, pad=1, optional bias, optional leaky-relu(0.2). grid.y = Cout.
// Border taps: clamped address + zeroed weight (unconditional loads).
// ---------------------------------------------------------------------------
__global__ void k_conv3x3(const float* __restrict__ x, const float* __restrict__ w,
                          const float* __restrict__ bias, float* __restrict__ y,
                          int Cin, int H, int wsh, int lrelu)
{
    const int Wd = 1 << wsh;
    const int npix = H << wsh;
    int p  = blockIdx.x * blockDim.x + threadIdx.x;
    int co = blockIdx.y;
    int py = p >> wsh, px = p & (Wd - 1);

    int   offs[9];
    float vmask[9];
    #pragma unroll
    for (int ky = 0; ky < 3; ++ky) {
        int yy = py + ky - 1;
        bool vy = (unsigned)yy < (unsigned)H;
        int yyc = vy ? yy : py;
        #pragma unroll
        for (int kx = 0; kx < 3; ++kx) {
            int xx = px + kx - 1;
            bool v = vy && ((unsigned)xx < (unsigned)Wd);
            int xxc = v ? xx : px;
            offs[ky * 3 + kx]  = (yyc << wsh) + xxc;
            vmask[ky * 3 + kx] = v ? 1.f : 0.f;
        }
    }

    float acc = bias ? bias[co] : 0.f;
    const float* wrow = w + (size_t)co * Cin * 9;
    for (int ci = 0; ci < Cin; ++ci) {
        const float* xc = x + (size_t)ci * npix;
        const float* wc = wrow + ci * 9;
        #pragma unroll
        for (int t = 0; t < 9; ++t)
            acc = fmaf(vmask[t] * wc[t], xc[offs[t]], acc);
    }
    if (lrelu) acc = (acc > 0.f) ? acc : 0.2f * acc;
    y[(size_t)co * npix + p] = acc;
}

// ---------------------------------------------------------------------------
// depthwise conv3x3, pad=1, no bias. grid.y = C. Same clamp+select scheme.
// ---------------------------------------------------------------------------
__global__ void k_dwconv3x3(const float* __restrict__ x, const float* __restrict__ w,
                            float* __restrict__ y, int H, int wsh)
{
    const int Wd = 1 << wsh;
    const int npix = H << wsh;
    int p = blockIdx.x * blockDim.x + threadIdx.x;
    int c = blockIdx.y;
    int py = p >> wsh, px = p & (Wd - 1);
    const float* xc = x + (size_t)c * npix;
    const float* wc = w + c * 9;
    float acc = 0.f;
    #pragma unroll
    for (int ky = 0; ky < 3; ++ky) {
        int yy = py + ky - 1;
        bool vy = (unsigned)yy < (unsigned)H;
        int yyc = vy ? yy : py;
        const float* xr = xc + ((size_t)yyc << wsh);
        #pragma unroll
        for (int kx = 0; kx < 3; ++kx) {
            int xx = px + kx - 1;
            bool v = vy && ((unsigned)xx < (unsigned)Wd);
            float wv = v ? wc[ky * 3 + kx] : 0.f;
            acc = fmaf(wv, xr[v ? xx : px], acc);
        }
    }
    y[(size_t)c * npix + p] = acc;
}

// ---------------------------------------------------------------------------
// Per-pixel LayerNorm over channels; writes zeros for pad rows [C, Cpad).
// ---------------------------------------------------------------------------
__global__ void k_layernorm(const float* __restrict__ x, const float* __restrict__ w,
                            const float* __restrict__ b, float* __restrict__ y,
                            int C, int Cpad, int npix)
{
    int p = blockIdx.x * blockDim.x + threadIdx.x;
    float mu = 0.f;
    for (int c = 0; c < C; ++c) mu += x[(size_t)c * npix + p];
    mu /= (float)C;
    float var = 0.f;
    for (int c = 0; c < C; ++c) { float d = x[(size_t)c * npix + p] - mu; var = fmaf(d, d, var); }
    var /= (float)C;
    float rs = rsqrtf(var + 1e-5f);
    for (int c = 0; c < C; ++c)
        y[(size_t)c * npix + p] = (x[(size_t)c * npix + p] - mu) * rs * w[c] + b[c];
    for (int c = C; c < Cpad; ++c)
        y[(size_t)c * npix + p] = 0.f;
}

// ---------------------------------------------------------------------------
// Zero floats (channel padding). Exact grid (count % 256 == 0).
// ---------------------------------------------------------------------------
__global__ void k_zero(float* __restrict__ p)
{
    size_t i = (size_t)blockIdx.x * blockDim.x + threadIdx.x;
    p[i] = 0.f;
}

// ---------------------------------------------------------------------------
// Pack weights [Cout,Cin] f32 -> bf16 WMMA A-fragments, zero-padded to
// MTpad M-tiles x KT K-tiles. Element i: e=i&15, lane=(i>>4)&31, tile=i>>9,
// kt=tile%KT, mt=tile/KT. k = kt*32 + 16*(lane>>4) + e (contiguous k order, same as the B loads).
// ---------------------------------------------------------------------------
__global__ void k_pack_w(const float* __restrict__ w, __bf16* __restrict__ pw,
                         int Cin, int Cout, int KT)
{
    int i = blockIdx.x * blockDim.x + threadIdx.x;
    int e    = i & 15;
    int lane = (i >> 4) & 31;
    int tile = i >> 9;
    int kt = tile % KT, mt = tile / KT;
    int hf = lane >> 4, l15 = lane & 15;
    int M = mt * 16 + l15;
    int K = kt * 32 + hf * 16 + e;
    float v = (M < Cout && K < Cin) ? w[(size_t)M * Cin + K] : 0.f;
    pw[i] = (__bf16)v;
}

// ---------------------------------------------------------------------------
// Pointwise conv GEMM: y[M,p] = sum_K w[M,K]*x[K,p] (+res), bf16 WMMA.
// NPIX compile-time => B loads use immediate offsets. Each wave: MTILE
// 16x16 M-tiles x 16 pixels. A fragments load as one clause; B fragment is
// double-buffered across K-tiles; prefetch covers kt+2. Host guarantees:
// NPIX%128==0, x rows zero-padded to KT*32 channels, weights packed for
// gridDim.y*MTILE tiles. EXEC stays all-ones (scalar-only branches).
// ---------------------------------------------------------------------------
template<int NPIX, int MTILE>
__global__ void k_pw_wmma_t(const float* __restrict__ x, const __bf16* __restrict__ pw,
                            const float* __restrict__ res, float* __restrict__ y,
                            int KT, int Cout)
{
    const int lane = threadIdx.x & 31;
    const int wid  = threadIdx.x >> 5;
    const int p0   = (blockIdx.x * 8 + wid) * 16;
    const int mt0  = blockIdx.y * MTILE;
    const int hf   = lane >> 4;
    const int l15  = lane & 15;
    const int pN   = p0 + l15;

    const v16bf* wt   = (const v16bf*)pw + ((size_t)mt0 * KT) * 32 + lane;
    const float* xcol = x + ((size_t)(hf * 16)) * NPIX + pN;

    v8f acc[MTILE];
    #pragma unroll
    for (int m = 0; m < MTILE; ++m)
        acc[m] = (v8f){0.f, 0.f, 0.f, 0.f, 0.f, 0.f, 0.f, 0.f};

    // prologue: first B fragment
    v16bf bv;
    #pragma unroll
    for (int e = 0; e < 16; ++e)
        bv[e] = (__bf16)xcol[(size_t)e * NPIX];          // immediate offsets

    for (int kt = 0; kt < KT; ++kt) {
        const float* xnext = xcol + (size_t)32 * NPIX;

        // A fragments for all M-tiles: one load clause (2x b128 per tile)
        v16bf av[MTILE];
        #pragma unroll
        for (int m = 0; m < MTILE; ++m)
            av[m] = wt[((size_t)m * KT + kt) * 32];

        // prefetch B two K-tiles ahead (global_prefetch_b8)
        if (kt + 2 < KT)
            __builtin_prefetch(xcol + (size_t)64 * NPIX, 0, 1);

        // next B fragment overlaps with the WMMA chain below
        v16bf bnext = bv;
        if (kt + 1 < KT) {
            #pragma unroll
            for (int e = 0; e < 16; ++e)
                bnext[e] = (__bf16)xnext[(size_t)e * NPIX];
        }

        #pragma unroll
        for (int m = 0; m < MTILE; ++m)
            acc[m] = __builtin_amdgcn_wmma_f32_16x16x32_bf16(
                false, av[m], false, bv, (short)0, acc[m], false, false);

        bv   = bnext;
        xcol = xnext;
    }

    #pragma unroll
    for (int m = 0; m < MTILE; ++m) {
        #pragma unroll
        for (int r = 0; r < 8; ++r) {
            int Mo = (mt0 + m) * 16 + hf * 8 + r;
            if (Mo < Cout) {
                float v = acc[m][r];
                if (res) v += res[(size_t)Mo * NPIX + pN];
                y[(size_t)Mo * NPIX + pN] = v;
            }
        }
    }
}

// ---------------------------------------------------------------------------
// Window channel-attention for one split. One block per window (256 threads).
// qkv: [3d, H, W]; split s uses channels [s*d,(s+1)*d): q|k|v each cc=d/3.
// Window is (1<<sh) x (1<<sh). cc <= 64, p <= 256.
// ---------------------------------------------------------------------------
__global__ void k_winattn(const float* __restrict__ qkv, float* __restrict__ ys,
                          int d, int H, int wsh, int split, int sh)
{
    __shared__ float s_attn[64 * 64];
    __shared__ float s_rq[64];
    __shared__ float s_rk[64];

    const int Wd   = 1 << wsh;
    const int dw   = 1 << sh;
    const int msk  = dw - 1;
    const int cc   = d / 3;
    const int p    = dw * dw;
    const int npix = H << wsh;
    const int nw   = Wd >> sh;
    const int wi   = blockIdx.x;
    const int gy0  = (wi / nw) << sh;
    const int gx0  = (wi % nw) << sh;
    const size_t base0 = ((size_t)gy0 << wsh) + gx0;
    const float* qb = qkv + (size_t)(split * d) * npix + base0;
    const float* kb = qb + (size_t)cc * npix;
    const float* vb = qb + (size_t)(2 * cc) * npix;
    const int tid = threadIdx.x;

    for (int c = tid; c < cc; c += blockDim.x) {
        const float* qr = qb + (size_t)c * npix;
        const float* kr = kb + (size_t)c * npix;
        float sq = 0.f, sk = 0.f;
        for (int t = 0; t < p; ++t) {
            int g = ((t >> sh) << wsh) + (t & msk);
            float a = qr[g], b = kr[g];
            sq = fmaf(a, a, sq);
            sk = fmaf(b, b, sk);
        }
        s_rq[c] = 1.f / fmaxf(sqrtf(sq), 1e-12f);
        s_rk[c] = 1.f / fmaxf(sqrtf(sk), 1e-12f);
    }
    __syncthreads();

    for (int idx = tid; idx < cc * cc; idx += blockDim.x) {
        int i = idx / cc, j = idx % cc;
        const float* qr = qb + (size_t)i * npix;
        const float* kr = kb + (size_t)j * npix;
        float s = 0.f;
        for (int t = 0; t < p; ++t) {
            int g = ((t >> sh) << wsh) + (t & msk);
            s = fmaf(qr[g], kr[g], s);
        }
        s_attn[idx] = s * s_rq[i] * s_rk[j];
    }
    __syncthreads();

    for (int i = tid; i < cc; i += blockDim.x) {
        float* row = s_attn + i * cc;
        float m = row[0];
        for (int j = 1; j < cc; ++j) m = fmaxf(m, row[j]);
        float sum = 0.f;
        for (int j = 0; j < cc; ++j) { float e = expf(row[j] - m); row[j] = e; sum += e; }
        float inv = 1.f / sum;
        for (int j = 0; j < cc; ++j) row[j] *= inv;
    }
    __syncthreads();

    float* yo = ys + (size_t)(split * cc) * npix + base0;
    for (int idx = tid; idx < cc * p; idx += blockDim.x) {
        int i = idx / p, t = idx % p;
        int g = ((t >> sh) << wsh) + (t & msk);
        const float* arow = s_attn + i * cc;
        float acc = 0.f;
        for (int j = 0; j < cc; ++j)
            acc = fmaf(arow[j], vb[(size_t)j * npix + g], acc);
        yo[(size_t)i * npix + g] = acc;
    }
}

// ---------------------------------------------------------------------------
// y[c] = gelu_exact(t[c]) * t[c+h].  grid.y = h. Exact grid over pixels.
// ---------------------------------------------------------------------------
__global__ void k_gelu_gate(const float* __restrict__ t, float* __restrict__ y,
                            int h, int npix)
{
    int p = blockIdx.x * blockDim.x + threadIdx.x;
    int c = blockIdx.y;
    float x1 = t[(size_t)c * npix + p];
    float x2 = t[(size_t)(c + h) * npix + p];
    float g = 0.5f * x1 * (1.f + erff(x1 * 0.70710678118654752f));
    y[(size_t)c * npix + p] = g * x2;
}

// ---------------------------------------------------------------------------
// pixel_unshuffle r=2: in [C,H,W] -> out [4C,H/2,W/2]. grid.y = 4C.
// ---------------------------------------------------------------------------
__global__ void k_unshuffle2(const float* __restrict__ x, float* __restrict__ y,
                             int H, int wsh)
{
    int Wo = 1 << (wsh - 1);
    int npo = (H >> 1) * Wo;
    int p  = blockIdx.x * blockDim.x + threadIdx.x;
    int co = blockIdx.y;
    int yo = p >> (wsh - 1), xo = p & (Wo - 1);
    int c = co >> 2, ry = (co >> 1) & 1, rx = co & 1;
    y[(size_t)co * npo + p] =
        x[((size_t)c * H << wsh) + ((size_t)(yo * 2 + ry) << wsh) + (xo * 2 + rx)];
}

// pixel_shuffle r=2: in [C,H,W] -> out [C/4,2H,2W]. grid.y = C/4.
__global__ void k_shuffle2(const float* __restrict__ x, float* __restrict__ y,
                           int H, int wsh)
{
    int Wo = 1 << (wsh + 1);
    int p  = blockIdx.x * blockDim.x + threadIdx.x;
    int co = blockIdx.y;
    int yo = p >> (wsh + 1), xo = p & (Wo - 1);
    int npo = (H * 2) * Wo;
    int ry = yo & 1, rx = xo & 1;
    y[(size_t)co * npo + p] =
        x[((size_t)(co * 4 + ry * 2 + rx) * H << wsh) + ((size_t)(yo >> 1) << wsh) + (xo >> 1)];
}

// ===========================================================================
// Host-side orchestration
// ===========================================================================
namespace {

struct BlockP {
    const float *n1w, *n1b, *qkv_w, *adw_w, *proj_w, *n2w, *n2b, *in_w, *fdw_w, *out_w;
};
struct Scratch { float *LNB, *SA, *SB, *SC; __bf16* PACKW; };

static inline int ffn_hidden(int d) { return (int)(d * 2.66); }
static inline int ceil32(int c) { return (c + 31) & ~31; }

template<int MTILE>
static void pw_dispatch(const float* x, const __bf16* pw, const float* res, float* y,
                        int KT, int Cout, int npix, int gy, hipStream_t s)
{
    dim3 grid(npix / 128, gy);
    switch (npix) {
    case 2048:   k_pw_wmma_t<2048,   MTILE><<<grid, 256, 0, s>>>(x, pw, res, y, KT, Cout); break;
    case 8192:   k_pw_wmma_t<8192,   MTILE><<<grid, 256, 0, s>>>(x, pw, res, y, KT, Cout); break;
    case 32768:  k_pw_wmma_t<32768,  MTILE><<<grid, 256, 0, s>>>(x, pw, res, y, KT, Cout); break;
    case 131072: k_pw_wmma_t<131072, MTILE><<<grid, 256, 0, s>>>(x, pw, res, y, KT, Cout); break;
    default: break;
    }
}

static void pw_conv(const float* x, const float* w, const float* res, float* y,
                    int Cin, int Cout, int npix, __bf16* packbuf, hipStream_t s)
{
    int MT = (Cout + 15) / 16;
    int KT = (Cin + 31) / 32;
    // Bandwidth-bound: prefer fewer activation passes (bigger MTILE) even if
    // it pads a few zero tiles - matrix throughput is not the bottleneck.
    int mtile = (MT >= 3) ? 4 : 2;
    int gy    = (MT + mtile - 1) / mtile;
    int MTpad = gy * mtile;
    int total = MTpad * KT * 512;
    k_pack_w<<<dim3(total / 256), 256, 0, s>>>(w, packbuf, Cin, Cout, KT);
    if (mtile == 4) pw_dispatch<4>(x, packbuf, res, y, KT, Cout, npix, gy, s);
    else            pw_dispatch<2>(x, packbuf, res, y, KT, Cout, npix, gy, s);
}

static void run_block(float* cur, int d, int H, int wsh, const BlockP& bp,
                      const Scratch& ws, hipStream_t s)
{
    const int npix = H << wsh;
    const int h  = ffn_hidden(d);
    const int dp = ceil32(d);
    const int hp = ceil32(h);
    dim3 gpix(npix / 256);

    // ---- x = x + attention(LN1(x)) ----
    k_layernorm<<<gpix, 256, 0, s>>>(cur, bp.n1w, bp.n1b, ws.LNB, d, dp, npix);
    pw_conv(ws.LNB, bp.qkv_w, nullptr, ws.SA, d, 3 * d, npix, ws.PACKW, s);
    k_dwconv3x3<<<dim3(gpix.x, 3 * d), 256, 0, s>>>(ws.SA, bp.adw_w, ws.SB, H, wsh);
    if (dp > d)
        k_zero<<<dim3((unsigned)(((size_t)(dp - d) * npix) / 256)), 256, 0, s>>>(
            ws.SC + (size_t)d * npix);
    for (int sp = 0; sp < 3; ++sp) {
        int sh = 2 + sp;                              // windows 4, 8, 16
        int nwin = (H >> sh) * ((1 << wsh) >> sh);
        k_winattn<<<dim3(nwin), 256, 0, s>>>(ws.SB, ws.SC, d, H, wsh, sp, sh);
    }
    pw_conv(ws.SC, bp.proj_w, cur, cur, d, d, npix, ws.PACKW, s);   // fused residual

    // ---- x = x + ffn(LN2(x)) ----
    k_layernorm<<<gpix, 256, 0, s>>>(cur, bp.n2w, bp.n2b, ws.LNB, d, dp, npix);
    pw_conv(ws.LNB, bp.in_w, nullptr, ws.SA, d, 2 * h, npix, ws.PACKW, s);
    k_dwconv3x3<<<dim3(gpix.x, 2 * h), 256, 0, s>>>(ws.SA, bp.fdw_w, ws.SB, H, wsh);
    k_gelu_gate<<<dim3(gpix.x, h), 256, 0, s>>>(ws.SB, ws.SC, h, npix);
    if (hp > h)
        k_zero<<<dim3((unsigned)(((size_t)(hp - h) * npix) / 256)), 256, 0, s>>>(
            ws.SC + (size_t)h * npix);
    pw_conv(ws.SC, bp.out_w, cur, cur, h, d, npix, ws.PACKW, s);    // fused residual
}

} // namespace

extern "C" void kernel_launch(void* const* d_in, const int* in_sizes, int n_in,
                              void* d_out, int out_size, void* d_ws, size_t ws_size,
                              hipStream_t stream)
{
    (void)in_sizes; (void)n_in; (void)out_size;

    // ---- parameter table: setup_inputs() insertion order, depth-first ----
    int idx = 0;
    auto P = [&]() -> const float* { return (const float*)d_in[idx++]; };
    const float* rgb = P();
    const float* dep = P();
    const float *rgb_w = P(), *rgb_b = P();
    const float *dep_w = P(), *dep_b = P();
    auto PB = [&]() {
        BlockP b;
        b.n1w = P(); b.n1b = P();
        b.qkv_w = P(); b.adw_w = P(); b.proj_w = P();
        b.n2w = P(); b.n2b = P();
        b.in_w = P(); b.fdw_w = P(); b.out_w = P();
        return b;
    };
    BlockP enc1[2]   = {PB(), PB()};
    const float* down1_w = P();
    BlockP enc2[2]   = {PB(), PB()};
    const float* down2_w = P();
    BlockP enc3[2]   = {PB(), PB()};
    const float* down3_w = P();
    BlockP latent[2] = {PB(), PB()};
    const float* up4_w  = P();
    const float* red3_w = P();
    BlockP dec3[2]   = {PB(), PB()};
    const float* up3_w  = P();
    const float* red2_w = P();
    BlockP dec2[2]   = {PB(), PB()};
    const float* up2_w  = P();
    BlockP dec1[2]   = {PB(), PB()};
    BlockP refine[2] = {PB(), PB()};
    const float* outw = P();

    // ---- geometry (all widths powers of two) ----
    const int H1 = 256, H2 = 128, H3 = 64, H4 = 32;
    const int w1 = 9, w2 = 8, w3 = 7, w4 = 6;        // log2(W) per level
    const size_t np1 = (size_t)H1 << w1, np2 = (size_t)H2 << w2,
                 np3 = (size_t)H3 << w3, np4 = (size_t)H4 << w4;

    // ---- workspace layout (floats) ----
    float* base = (float*)d_ws;
    size_t off = 0;
    auto A = [&](size_t n) { float* r = base + off; off += (n + 255) & ~(size_t)255; return r; };
    float* X0   = A(24 * np1);            // stem features (refine skip)
    float* E1   = A(24 * np1);            // enc1 output (dec1 skip)
    float* E2   = A(48 * np2);            // enc2 output (dec2 skip)
    float* E3   = A(96 * np3);            // enc3 output (dec3 skip)
    float* CUR  = A(48 * np2);            // latent / dec3 / dec2 current
    float* CAT  = A(72 * np1);            // concat buffers; dec1+refine current
    float* TMPA = A(96 * np2);            // conv3x3 outputs pre-(un)shuffle
    Scratch ws;
    ws.LNB   = A(96 * np1);               // LN output, padded to ceil32(d)
    ws.SA    = A((size_t)382 * np1);      // qkv / ffn-hidden (max(3d,2h)*npix)
    ws.SB    = A((size_t)382 * np1);      // after depthwise conv
    ws.SC    = A((size_t)192 * np1);      // attention ys / gelu gate, padded
    ws.PACKW = (__bf16*)A(1 << 19);       // packed bf16 weights (1M elems)
    if (off * sizeof(float) > ws_size) return;   // deterministic bail if ws too small

    // ---- stem: x0 = concat(lrelu(conv(rgb)), lrelu(conv(dep))) ----
    k_conv3x3<<<dim3(np1 / 256, 18), 256, 0, stream>>>(rgb, rgb_w, rgb_b, X0,            3, H1, w1, 1);
    k_conv3x3<<<dim3(np1 / 256,  6), 256, 0, stream>>>(dep, dep_w, dep_b, X0 + 18 * np1, 1, H1, w1, 1);

    // ---- encoder level 1 ----
    hipMemcpyAsync(E1, X0, 24 * np1 * sizeof(float), hipMemcpyDeviceToDevice, stream);
    for (int i = 0; i < 2; ++i) run_block(E1, 24, H1, w1, enc1[i], ws, stream);

    // ---- down1 -> level 2 ----
    k_conv3x3<<<dim3(np1 / 256, 12), 256, 0, stream>>>(E1, down1_w, nullptr, TMPA, 24, H1, w1, 0);
    k_unshuffle2<<<dim3(np2 / 256, 48), 256, 0, stream>>>(TMPA, E2, H1, w1);
    for (int i = 0; i < 2; ++i) run_block(E2, 48, H2, w2, enc2[i], ws, stream);

    // ---- down2 -> level 3 ----
    k_conv3x3<<<dim3(np2 / 256, 24), 256, 0, stream>>>(E2, down2_w, nullptr, TMPA, 48, H2, w2, 0);
    k_unshuffle2<<<dim3(np3 / 256, 96), 256, 0, stream>>>(TMPA, E3, H2, w2);
    for (int i = 0; i < 2; ++i) run_block(E3, 96, H3, w3, enc3[i], ws, stream);

    // ---- down3 -> latent ----
    k_conv3x3<<<dim3(np3 / 256, 48), 256, 0, stream>>>(E3, down3_w, nullptr, TMPA, 96, H3, w3, 0);
    k_unshuffle2<<<dim3(np4 / 256, 192), 256, 0, stream>>>(TMPA, CUR, H3, w3);
    for (int i = 0; i < 2; ++i) run_block(CUR, 192, H4, w4, latent[i], ws, stream);

    // ---- up4 -> dec3: d3 = red3(concat(shuffle(conv(lat)), e3)) ----
    k_conv3x3<<<dim3(np4 / 256, 384), 256, 0, stream>>>(CUR, up4_w, nullptr, TMPA, 192, H4, w4, 0);
    k_shuffle2<<<dim3(np3 / 256, 96), 256, 0, stream>>>(TMPA, CAT, H4, w4);
    hipMemcpyAsync(CAT + 96 * np3, E3, 96 * np3 * sizeof(float), hipMemcpyDeviceToDevice, stream);
    pw_conv(CAT, red3_w, nullptr, CUR, 192, 96, (int)np3, ws.PACKW, stream);
    for (int i = 0; i < 2; ++i) run_block(CUR, 96, H3, w3, dec3[i], ws, stream);

    // ---- up3 -> dec2 ----
    k_conv3x3<<<dim3(np3 / 256, 192), 256, 0, stream>>>(CUR, up3_w, nullptr, TMPA, 96, H3, w3, 0);
    k_shuffle2<<<dim3(np2 / 256, 48), 256, 0, stream>>>(TMPA, CAT, H3, w3);
    hipMemcpyAsync(CAT + 48 * np2, E2, 48 * np2 * sizeof(float), hipMemcpyDeviceToDevice, stream);
    pw_conv(CAT, red2_w, nullptr, CUR, 96, 48, (int)np2, ws.PACKW, stream);
    for (int i = 0; i < 2; ++i) run_block(CUR, 48, H2, w2, dec2[i], ws, stream);

    // ---- up2 -> dec1 (no channel reduce, Restormer style) ----
    k_conv3x3<<<dim3(np2 / 256, 96), 256, 0, stream>>>(CUR, up2_w, nullptr, TMPA, 48, H2, w2, 0);
    k_shuffle2<<<dim3(np1 / 256, 24), 256, 0, stream>>>(TMPA, CAT, H2, w2);
    hipMemcpyAsync(CAT + 24 * np1, E1, 24 * np1 * sizeof(float), hipMemcpyDeviceToDevice, stream);
    for (int i = 0; i < 2; ++i) run_block(CAT, 48, H1, w1, dec1[i], ws, stream);

    // ---- refine: concat(d1, x0) -> 72 channels ----
    hipMemcpyAsync(CAT + 48 * np1, X0, 24 * np1 * sizeof(float), hipMemcpyDeviceToDevice, stream);
    for (int i = 0; i < 2; ++i) run_block(CAT, 72, H1, w1, refine[i], ws, stream);

    // ---- output conv3x3: 72 -> 1 ----
    k_conv3x3<<<dim3(np1 / 256, 1), 256, 0, stream>>>(CAT, outw, nullptr, (float*)d_out, 72, H1, w1, 0);
}
